// FDM_46394236731667
// MI455X (gfx1250) — matrix operand order repaired and checked by reading, not compiled, not run on hardware
//
#include <hip/hip_runtime.h>
#include <stdint.h>

// Problem constants (reference: B,C,H,W = 8,512,56,56)
#define BATCH  8
#define CH     512
#define NTOK   3136          // 56*56
#define NPAD   3200          // 25 * 128, zero-padded keys
#define QT     16            // queries per workgroup (one 16x16 WMMA tile of rows)
#define KT     128           // keys per key-tile (8 waves x 16 keys)
#define NKT    (NPAD / KT)   // 25
#define FACTOR 0.001f

typedef __attribute__((ext_vector_type(16))) __bf16 v16bf;
typedef __attribute__((ext_vector_type(8)))  float  v8f;

union FragU {
    uint4 q[2];   // 32 bytes
    v16bf bf;     // 16 bf16 = 32 bytes
};

static __device__ __forceinline__ unsigned short f2bf(float f) {
    // round-to-nearest-even f32 -> bf16
    unsigned int u = __float_as_uint(f);
    u += 0x7FFFu + ((u >> 16) & 1u);
    return (unsigned short)(u >> 16);
}

// ---------------------------------------------------------------------------
// Prep 1: per-column inverse L2 norms over channels for all three inputs.
// ---------------------------------------------------------------------------
__global__ void fdm_norms(const float* __restrict__ f1, const float* __restrict__ f2,
                          const float* __restrict__ f3, float* __restrict__ rn1,
                          float* __restrict__ rn2, float* __restrict__ rn3) {
    int idx = blockIdx.x * blockDim.x + threadIdx.x;
    if (idx >= BATCH * NTOK) return;
    size_t base = (size_t)(idx / NTOK) * CH * NTOK + (size_t)(idx % NTOK);
    float s1 = 0.f, s2 = 0.f, s3 = 0.f;
    for (int c = 0; c < CH; ++c) {
        size_t o = base + (size_t)c * NTOK;
        float a = f1[o], b = f2[o], d = f3[o];
        s1 += a * a; s2 += b * b; s3 += d * d;
    }
    rn1[idx] = 1.f / fmaxf(sqrtf(s1), 1e-12f);
    rn2[idx] = 1.f / fmaxf(sqrtf(s2), 1e-12f);
    rn3[idx] = 1.f / fmaxf(sqrtf(s3), 1e-12f);
}

// ---------------------------------------------------------------------------
// Prep 2: normalized Q [b][n][c] bf16, normalized K2/K3 [b][npad][c] bf16
//         (zero rows for padded keys).
// ---------------------------------------------------------------------------
__global__ void fdm_pack_qk(const float* __restrict__ f1, const float* __restrict__ f2,
                            const float* __restrict__ f3, const float* __restrict__ rn1,
                            const float* __restrict__ rn2, const float* __restrict__ rn3,
                            unsigned short* __restrict__ Q, unsigned short* __restrict__ K2,
                            unsigned short* __restrict__ K3) {
    int idx = blockIdx.x * blockDim.x + threadIdx.x;
    if (idx >= BATCH * NPAD) return;
    int b = idx / NPAD, n = idx % NPAD;
    unsigned short* k2p = K2 + (size_t)idx * CH;
    unsigned short* k3p = K3 + (size_t)idx * CH;
    if (n < NTOK) {
        size_t src = (size_t)b * CH * NTOK + n;
        float r1 = rn1[b * NTOK + n];
        float r2 = rn2[b * NTOK + n];
        float r3 = rn3[b * NTOK + n];
        unsigned short* qp = Q + ((size_t)b * NTOK + n) * CH;
        for (int c = 0; c < CH; ++c) {
            size_t o = src + (size_t)c * NTOK;
            qp[c]  = f2bf(f1[o] * r1);
            k2p[c] = f2bf(f2[o] * r2);
            k3p[c] = f2bf(f3[o] * r3);
        }
    } else {
        for (int c = 0; c < CH; ++c) { k2p[c] = 0; k3p[c] = 0; }
    }
}

// ---------------------------------------------------------------------------
// Prep 3: raw V2/V3 as bf16 in [b][c][npad] (zero for padded keys).
// ---------------------------------------------------------------------------
__global__ void fdm_pack_v(const float* __restrict__ f2, const float* __restrict__ f3,
                           unsigned short* __restrict__ V2, unsigned short* __restrict__ V3) {
    size_t idx = (size_t)blockIdx.x * blockDim.x + threadIdx.x;
    if (idx >= (size_t)BATCH * CH * NPAD) return;
    size_t bc = idx / NPAD;
    int n = (int)(idx % NPAD);
    unsigned short v2 = 0, v3 = 0;
    if (n < NTOK) {
        size_t o = bc * NTOK + (size_t)n;
        v2 = f2bf(f2[o]);
        v3 = f2bf(f3[o]);
    }
    V2[idx] = v2;
    V3[idx] = v3;
}

// ---------------------------------------------------------------------------
// Main fused attention: one workgroup = (batch b, 16-query tile).
// Scores bounded in [-1,1] (unit vectors) => exp without max-tracking.
// ---------------------------------------------------------------------------
__global__ __launch_bounds__(256)
void fdm_attn(const float* __restrict__ fm1,
              const unsigned short* __restrict__ Qbf,
              const unsigned short* __restrict__ K2,
              const unsigned short* __restrict__ K3,
              const unsigned short* __restrict__ V2,
              const unsigned short* __restrict__ V3,
              float* __restrict__ out) {
    __shared__ __align__(16) unsigned short sQ[QT * CH];      // 16 KB, Q tile [q][c]
    __shared__ __align__(16) unsigned short sP[2][QT][KT];    // 8 KB, exp probs [branch][q][m]
    __shared__ float sL[2][QT];                               // softmax denominators

    const int tid  = threadIdx.x;
    const int wave = tid >> 5;
    const int lane = tid & 31;
    const int r    = lane & 15;   // row within 16-lane half
    const int h    = lane >> 4;   // half select
    const int b    = blockIdx.y;
    const int n0   = blockIdx.x * QT;

    // Stage Q tile: 16 rows x 512 bf16, contiguous in [b][n][c] layout.
    {
        const uint4* src = (const uint4*)(Qbf + ((size_t)b * NTOK + n0) * CH);
        uint4* dst = (uint4*)sQ;
        for (int i = tid; i < QT * CH / 8; i += 256) dst[i] = src[i];
    }
    if (tid < 2 * QT) ((float*)sL)[tid] = 0.f;
    __syncthreads();

    const unsigned short* Kp[2] = { K2, K3 };
    const unsigned short* Vp[2] = { V2, V3 };

    v8f acc[2][4];   // 2 branches x 4 channel sub-tiles (wave owns 64 channels)
    #pragma unroll
    for (int i = 0; i < 2; ++i)
        #pragma unroll
        for (int j = 0; j < 4; ++j) { v8f z = {}; acc[i][j] = z; }

    for (int mt = 0; mt < NKT; ++mt) {
        const int m0 = mt * KT + wave * QT;           // this wave's 16 keys
        const bool kvalid = (m0 + r) < NTOK;

        if (mt + 1 < NKT) {  // prefetch next key tile rows (global_prefetch_b8)
            size_t nrow = ((size_t)(b * NPAD + (mt + 1) * KT + wave * QT + r)) * CH;
            __builtin_prefetch(Kp[0] + nrow, 0, 1);
            __builtin_prefetch(Kp[1] + nrow, 0, 1);
        }

        // ---- Phase 1: S = Qn^T Kn over C=512 (16 WMMAs per branch per wave) ----
        #pragma unroll
        for (int br = 0; br < 2; ++br) {
            const unsigned short* kbase =
                Kp[br] + ((size_t)(b * NPAD + m0 + r)) * CH + h * 8;
            v8f s = {};
            #pragma unroll
            for (int kc = 0; kc < 16; ++kc) {
                const int c0 = kc * 32;
                FragU a, bb;
                a.q[0] = *(const uint4*)(sQ + r * CH + c0 + h * 8);
                a.q[1] = *(const uint4*)(sQ + r * CH + c0 + 16 + h * 8);
                const uint4* kb = (const uint4*)(kbase + c0);
                bb.q[0] = kb[0];
                bb.q[1] = kb[2];
                s = __builtin_amdgcn_wmma_f32_16x16x32_bf16(
                        false, a.bf, false, bb.bf, (short)0, s, false, false);
            }
            // D layout: lane -> key m = r, query q = h*8+v. p = exp(-dot).
            float pv[8], rs[8];
            #pragma unroll
            for (int v = 0; v < 8; ++v) {
                float p = __expf(-s[v]);
                p = kvalid ? p : 0.f;
                pv[v] = p;
                rs[v] = p;
            }
            // Row-sum over the 16 keys held across lanes of each half.
            #pragma unroll
            for (int mask = 1; mask < 16; mask <<= 1)
                #pragma unroll
                for (int v = 0; v < 8; ++v) rs[v] += __shfl_xor(rs[v], mask, 32);
            if ((lane & 15) == 0) {
                #pragma unroll
                for (int v = 0; v < 8; ++v) atomicAdd(&sL[br][h * 8 + v], rs[v]);
            }
            #pragma unroll
            for (int v = 0; v < 8; ++v)
                sP[br][h * 8 + v][wave * QT + r] = f2bf(pv[v]);
        }
        __syncthreads();

        // ---- Phase 2: acc[c,q] += V[c,m] * P[q,m]; wave owns channels [wave*64,+64) ----
        #pragma unroll
        for (int br = 0; br < 2; ++br) {
            #pragma unroll
            for (int ct = 0; ct < 4; ++ct) {
                const int crow = wave * 64 + ct * 16 + r;
                const unsigned short* vbase =
                    Vp[br] + ((size_t)(b * CH + crow)) * NPAD + mt * KT;
                #pragma unroll
                for (int kk = 0; kk < 4; ++kk) {
                    FragU a, bb;
                    a.q[0] = *(const uint4*)(vbase + kk * 32 + h * 8);
                    a.q[1] = *(const uint4*)(vbase + kk * 32 + 16 + h * 8);
                    const uint4* pb = (const uint4*)(&sP[br][r][kk * 32 + h * 8]);
                    bb.q[0] = pb[0];
                    bb.q[1] = pb[2];
                    acc[br][ct] = __builtin_amdgcn_wmma_f32_16x16x32_bf16(
                        false, a.bf, false, bb.bf, (short)0, acc[br][ct], false, false);
                }
            }
        }
        __syncthreads();  // protect sP before next tile's Phase 1
    }

    __syncthreads();
    const float i2 = FACTOR / sL[0][r];
    const float i3 = FACTOR / sL[1][r];
    const int n = n0 + r;   // D layout: lane -> query q = r (N index), channel = h*8+v (M index)
    #pragma unroll
    for (int ct = 0; ct < 4; ++ct) {
        #pragma unroll
        for (int v = 0; v < 8; ++v) {
            const int c = wave * 64 + ct * 16 + h * 8 + v;
            size_t o = ((size_t)b * CH + c) * NTOK + n;
            out[o] = fm1[o] + i2 * acc[0][ct][v] + i3 * acc[1][ct][v];
        }
    }
}

// ---------------------------------------------------------------------------
extern "C" void kernel_launch(void* const* d_in, const int* in_sizes, int n_in,
                              void* d_out, int out_size, void* d_ws, size_t ws_size,
                              hipStream_t stream) {
    const float* fm1 = (const float*)d_in[0];
    const float* fm2 = (const float*)d_in[1];
    const float* fm3 = (const float*)d_in[2];
    float* out = (float*)d_out;

    char* ws = (char*)d_ws;
    size_t off = 0;
    auto take = [&](size_t bytes) -> char* {
        char* p = ws + off;
        off += (bytes + 255) & ~(size_t)255;
        return p;
    };
    float* rn1 = (float*)take((size_t)BATCH * NTOK * 4);
    float* rn2 = (float*)take((size_t)BATCH * NTOK * 4);
    float* rn3 = (float*)take((size_t)BATCH * NTOK * 4);
    unsigned short* Qbf = (unsigned short*)take((size_t)BATCH * NTOK * CH * 2);
    unsigned short* K2  = (unsigned short*)take((size_t)BATCH * NPAD * CH * 2);
    unsigned short* K3  = (unsigned short*)take((size_t)BATCH * NPAD * CH * 2);
    unsigned short* V2  = (unsigned short*)take((size_t)BATCH * CH * NPAD * 2);
    unsigned short* V3  = (unsigned short*)take((size_t)BATCH * CH * NPAD * 2);

    fdm_norms<<<(BATCH * NTOK + 255) / 256, 256, 0, stream>>>(fm1, fm2, fm3, rn1, rn2, rn3);
    fdm_pack_qk<<<(BATCH * NPAD + 255) / 256, 256, 0, stream>>>(
        fm1, fm2, fm3, rn1, rn2, rn3, Qbf, K2, K3);
    size_t nv = (size_t)BATCH * CH * NPAD;
    fdm_pack_v<<<(unsigned)((nv + 255) / 256), 256, 0, stream>>>(fm2, fm3, V2, V3);

    dim3 grid(NTOK / QT, BATCH);   // 196 query tiles x 8 batches
    fdm_attn<<<grid, 256, 0, stream>>>(fm1, Qbf, K2, K3, V2, V3, out);
}
